// GraphKmTwoStreamEdgeNet_75110388073226
// MI455X (gfx1250) — matrix operand order repaired and checked by reading, not compiled, not run on hardware
//
#include <hip/hip_runtime.h>
#include <hip/hip_bf16.h>

typedef __attribute__((ext_vector_type(16))) _Float16 v16h;
typedef __attribute__((ext_vector_type(8)))  float    v8f;

#define BN_EPS 1e-5f

// ---------------------------------------------------------------------------
// WMMA helpers (CDNA5, wave32, 16x16x32 f16 -> f32 accumulate)
// ---------------------------------------------------------------------------
__device__ __forceinline__ v8f wmma_f16(v16h a, v16h b, v8f c) {
  // 8 args: (neg_a, A, neg_b, B, c_mod, C, reuse_a, reuse_b)
  return __builtin_amdgcn_wmma_f32_16x16x32_f16(false, a, false, b, (short)0, c,
                                                false, false);
}

// A fragment from an f16 row-major tile in LDS (16 rows x ldk cols).
// ISA 16-bit A layout: lane<16 -> row=lane, K {k0..k0+7, k0+16..k0+23}
//                      lane>=16 -> row=lane-16, K {k0+8..k0+15, k0+24..k0+31}
__device__ __forceinline__ v16h lds_a_frag(const _Float16* lds, int ldk, int k0,
                                           int lane) {
  int row = lane & 15;
  const _Float16* p = lds + row * ldk + k0 + ((lane >> 4) << 3);
  v16h a;
#pragma unroll
  for (int j = 0; j < 8; ++j) { a[j] = p[j]; a[8 + j] = p[16 + j]; }
  return a;
}

// B fragment from pre-packed weights (fragment-order, see pack_b_kernel).
__device__ __forceinline__ v16h load_b(const _Float16* Bp, int kt, int nt,
                                       int numKt, int lane) {
  const v16h* p = (const v16h*)Bp;
  return p[((long)nt * numKt + kt) * 32 + lane];
}

// Pack f32 row-major W[K][N] into per-(kt,nt)-tile fragment order:
// ISA 16-bit B layout: lane<16 -> col = nt*16+lane, K {kt*32+0..7, kt*32+16..23}
//                      lane>=16 -> col = nt*16+lane-16, K {kt*32+8..15, kt*32+24..31}
__global__ void pack_b_kernel(const float* __restrict__ W,
                              _Float16* __restrict__ Bp, int K, int Ncols) {
  int numKt = (K + 31) >> 5;
  int numNt = (Ncols + 15) >> 4;
  long total = (long)numKt * numNt * 512;
  long idx = (long)blockIdx.x * blockDim.x + threadIdx.x;
  if (idx >= total) return;
  int j = (int)(idx & 15);
  int lane = (int)((idx >> 4) & 31);
  long tile = idx >> 9;
  int kt = (int)(tile % numKt);
  int nt = (int)(tile / numKt);
  int col = nt * 16 + (lane & 15);
  int k = kt * 32 + ((lane >> 4) << 3) + (j & 7) + ((j >> 3) << 4);
  float v = (k < K && col < Ncols) ? W[(long)k * Ncols + col] : 0.f;
  Bp[idx] = (_Float16)v;
}

// ---------------------------------------------------------------------------
// Layer 0: gf[rows in [lo,hi)] = relu(bn(X_slice @ W + b)); one wave = 16x16 tile
// ---------------------------------------------------------------------------
__global__ void gemm_layer0_kernel(const float* __restrict__ X, int xld, int coloff,
                                   const _Float16* __restrict__ Bp, int K,
                                   const float* __restrict__ bias,
                                   const float* __restrict__ bg, const float* __restrict__ bb,
                                   const float* __restrict__ bm, const float* __restrict__ bv,
                                   float* __restrict__ Out,
                                   const int* __restrict__ na_ptr, int sel, int Nrows) {
  int lane = threadIdx.x;
  int na = *na_ptr;
  int lo = (sel == 0) ? 0 : na;
  int hi = (sel == 0) ? na : Nrows;
  int r0 = blockIdx.x * 16;
  int nt = blockIdx.y;
  if (r0 + 16 <= lo || r0 >= hi) return;
  int numKt = K >> 5;
  int arow = r0 + (lane & 15);
  if (arow >= Nrows) arow = Nrows - 1;
  const float* ap = X + (long)arow * xld + coloff + ((lane >> 4) << 3);
  v8f acc = {0.f, 0.f, 0.f, 0.f, 0.f, 0.f, 0.f, 0.f};
  for (int kt = 0; kt < numKt; ++kt) {
    const float* p = ap + kt * 32;
    v16h a;
#pragma unroll
    for (int j = 0; j < 8; ++j) { a[j] = (_Float16)p[j]; a[8 + j] = (_Float16)p[16 + j]; }
    v16h b = load_b(Bp, kt, nt, numKt, lane);
    acc = wmma_f16(a, b, acc);
  }
  int col = nt * 16 + (lane & 15);
  float gs = bg[col] / sqrtf(bv[col] + BN_EPS);
  float mm = bm[col], be = bb[col], bi = bias[col];
#pragma unroll
  for (int r = 0; r < 8; ++r) {
    int row = r0 + r + ((lane >> 4) << 3);
    if (row >= lo && row < hi) {
      float v = (acc[r] + bi - mm) * gs + be;
      Out[(long)row * 64 + col] = fmaxf(v, 0.f);
    }
  }
}

// Generic: Out[Nrows x Ncols] = A[Nrows x K] @ W + bias   (xh projection)
__global__ void gemm_bias_kernel(const float* __restrict__ A, int lda,
                                 const _Float16* __restrict__ Bp, int K, int Ncols,
                                 const float* __restrict__ bias,
                                 float* __restrict__ Out, int Nrows) {
  int lane = threadIdx.x;
  int r0 = blockIdx.x * 16;
  int nt = blockIdx.y;
  int numKt = K >> 5;
  int arow = r0 + (lane & 15);
  if (arow >= Nrows) arow = Nrows - 1;
  const float* ap = A + (long)arow * lda + ((lane >> 4) << 3);
  v8f acc = {0.f, 0.f, 0.f, 0.f, 0.f, 0.f, 0.f, 0.f};
  for (int kt = 0; kt < numKt; ++kt) {
    const float* p = ap + kt * 32;
    v16h a;
#pragma unroll
    for (int j = 0; j < 8; ++j) { a[j] = (_Float16)p[j]; a[8 + j] = (_Float16)p[16 + j]; }
    v16h b = load_b(Bp, kt, nt, numKt, lane);
    acc = wmma_f16(a, b, acc);
  }
  int col = nt * 16 + (lane & 15);
  float bi = bias[col];
#pragma unroll
  for (int r = 0; r < 8; ++r) {
    int row = r0 + r + ((lane >> 4) << 3);
    if (row < Nrows) Out[(long)row * Ncols + col] = acc[r] + bi;
  }
}

// ---------------------------------------------------------------------------
// Fused LinearPathPreact over 16-row tiles (one wave):
//   lds <- relu(bn1(input row)) f16; WMMA fc1; relu(bn2(.)) -> lds; WMMA fc2.
// KmConv variant: input row = [x[n], aggr_mean[n]]; output = +fc2_b+km_bias, store.
// ---------------------------------------------------------------------------
__global__ void km_lpp_kernel(const float* __restrict__ Xin, const float* __restrict__ Aggr,
    const float* b1g, const float* b1b, const float* b1m, const float* b1v,
    const _Float16* __restrict__ f1p, const float* __restrict__ f1b,
    const float* b2g, const float* b2b, const float* b2m, const float* b2v,
    const _Float16* __restrict__ f2p, const float* __restrict__ f2b,
    const float* __restrict__ kmbias, float* __restrict__ Out, int Nrows) {
  __shared__ _Float16 ldsA[16 * 128];
  __shared__ _Float16 ldsH[16 * 64];
  int lane = threadIdx.x;
  int r0 = blockIdx.x * 16;
  for (int t = 0; t < 64; ++t) {
    int idx = t * 32 + lane;
    int row = idx >> 7, col = idx & 127;
    int gr = r0 + row; if (gr >= Nrows) gr = Nrows - 1;
    float v = (col < 64) ? Xin[(long)gr * 64 + col] : Aggr[(long)gr * 64 + (col - 64)];
    v = (v - b1m[col]) * (b1g[col] / sqrtf(b1v[col] + BN_EPS)) + b1b[col];
    ldsA[idx] = (_Float16)fmaxf(v, 0.f);
  }
  __syncthreads();
#pragma unroll
  for (int nt = 0; nt < 4; ++nt) {
    v8f acc = {0.f, 0.f, 0.f, 0.f, 0.f, 0.f, 0.f, 0.f};
#pragma unroll
    for (int kt = 0; kt < 4; ++kt)
      acc = wmma_f16(lds_a_frag(ldsA, 128, kt * 32, lane), load_b(f1p, kt, nt, 4, lane), acc);
    int col = nt * 16 + (lane & 15);
    float gs = b2g[col] / sqrtf(b2v[col] + BN_EPS);
    float mm = b2m[col], be = b2b[col], bi = f1b[col];
#pragma unroll
    for (int r = 0; r < 8; ++r) {
      int row = ((lane >> 4) << 3) + r;
      float v = (acc[r] + bi - mm) * gs + be;
      ldsH[row * 64 + col] = (_Float16)fmaxf(v, 0.f);
    }
  }
  __syncthreads();
#pragma unroll
  for (int nt = 0; nt < 4; ++nt) {
    v8f acc = {0.f, 0.f, 0.f, 0.f, 0.f, 0.f, 0.f, 0.f};
#pragma unroll
    for (int kt = 0; kt < 2; ++kt)
      acc = wmma_f16(lds_a_frag(ldsH, 64, kt * 32, lane), load_b(f2p, kt, nt, 2, lane), acc);
    int col = nt * 16 + (lane & 15);
    float ex = f2b[col] + kmbias[col];
#pragma unroll
    for (int r = 0; r < 8; ++r) {
      int gr = r0 + ((lane >> 4) << 3) + r;
      if (gr < Nrows) Out[(long)gr * 64 + col] = acc[r] + ex;
    }
  }
}

__device__ __forceinline__ bool edge_mask(int mode, int delta, int selfe) {
  bool se = (selfe == 1);
  switch (mode) {
    case 1: return delta < 1;
    case 2: return (delta >= 1 && delta < 4) || se;
    case 3: return (delta >= 4 && delta < 8) || se;
    case 4: return (delta >= 8 && delta < 15) || se;
    default: return delta >= 15;
  }
}

// EdgeConv variant: input row e = [x[dst], x[src]-x[dst]]; masked atomic scatter.
__global__ void ec_lpp_kernel(const float* __restrict__ Xin,
    const int* __restrict__ src, const int* __restrict__ dst,
    const int* __restrict__ edelta, const int* __restrict__ eself, int mode,
    const float* b1g, const float* b1b, const float* b1m, const float* b1v,
    const _Float16* __restrict__ f1p, const float* __restrict__ f1b,
    const float* b2g, const float* b2b, const float* b2m, const float* b2v,
    const _Float16* __restrict__ f2p, const float* __restrict__ f2b,
    float* __restrict__ Sacc, int E) {
  __shared__ _Float16 ldsA[16 * 128];
  __shared__ _Float16 ldsH[16 * 64];
  int lane = threadIdx.x;
  int e0 = blockIdx.x * 16;
  for (int t = 0; t < 64; ++t) {
    int idx = t * 32 + lane;
    int row = idx >> 7, col = idx & 127;
    int e = e0 + row; if (e >= E) e = E - 1;
    int d = dst[e];
    float v;
    if (col < 64) v = Xin[(long)d * 64 + col];
    else {
      int s = src[e]; int c = col - 64;
      v = Xin[(long)s * 64 + c] - Xin[(long)d * 64 + c];
    }
    v = (v - b1m[col]) * (b1g[col] / sqrtf(b1v[col] + BN_EPS)) + b1b[col];
    ldsA[idx] = (_Float16)fmaxf(v, 0.f);
  }
  __syncthreads();
#pragma unroll
  for (int nt = 0; nt < 4; ++nt) {
    v8f acc = {0.f, 0.f, 0.f, 0.f, 0.f, 0.f, 0.f, 0.f};
#pragma unroll
    for (int kt = 0; kt < 4; ++kt)
      acc = wmma_f16(lds_a_frag(ldsA, 128, kt * 32, lane), load_b(f1p, kt, nt, 4, lane), acc);
    int col = nt * 16 + (lane & 15);
    float gs = b2g[col] / sqrtf(b2v[col] + BN_EPS);
    float mm = b2m[col], be = b2b[col], bi = f1b[col];
#pragma unroll
    for (int r = 0; r < 8; ++r) {
      int row = ((lane >> 4) << 3) + r;
      float v = (acc[r] + bi - mm) * gs + be;
      ldsH[row * 64 + col] = (_Float16)fmaxf(v, 0.f);
    }
  }
  __syncthreads();
#pragma unroll
  for (int nt = 0; nt < 4; ++nt) {
    v8f acc = {0.f, 0.f, 0.f, 0.f, 0.f, 0.f, 0.f, 0.f};
#pragma unroll
    for (int kt = 0; kt < 2; ++kt)
      acc = wmma_f16(lds_a_frag(ldsH, 64, kt * 32, lane), load_b(f2p, kt, nt, 2, lane), acc);
    int col = nt * 16 + (lane & 15);
    float bi = f2b[col];
#pragma unroll
    for (int r = 0; r < 8; ++r) {
      int e = e0 + ((lane >> 4) << 3) + r;
      if (e < E && edge_mask(mode, edelta[e], eself[e]))
        atomicAdd(&Sacc[(long)dst[e] * 64 + col], acc[r] + bi);
    }
  }
}

// ---------------------------------------------------------------------------
// Attention (masked segment softmax) support kernels
// ---------------------------------------------------------------------------
__global__ void s_nodes_kernel(const float* __restrict__ xh,
                               const float* __restrict__ as_, const float* __restrict__ ad_,
                               float* __restrict__ ssrc, float* __restrict__ sdst, int N) {
  int i = blockIdx.x * blockDim.x + threadIdx.x;
  if (i >= N * 4) return;
  int n = i >> 2, h = i & 3;
  const float* xp = xh + (long)n * 256 + h * 64;
  const float* s_ = as_ + h * 64;
  const float* d_ = ad_ + h * 64;
  float s1 = 0.f, s2 = 0.f;
  for (int c = 0; c < 64; ++c) { float v = xp[c]; s1 += v * s_[c]; s2 += v * d_[c]; }
  ssrc[i] = s1; sdst[i] = s2;
}

// Reduce lin_e_w (16x256) against att_edge (4x64): w_eh[4][16], b_eh[4]
__global__ void prep_sedge_kernel(const float* __restrict__ lew, const float* __restrict__ leb,
                                  const float* __restrict__ ae, float* __restrict__ weh) {
  int i = threadIdx.x;  // 64 threads
  int h = i >> 4, col = i & 15;
  float s = 0.f;
  for (int c = 0; c < 64; ++c) s += lew[col * 256 + h * 64 + c] * ae[h * 64 + c];
  weh[i] = s;
  if (col == 0) {
    float sb = 0.f;
    for (int c = 0; c < 64; ++c) sb += leb[h * 64 + c] * ae[h * 64 + c];
    weh[64 + h] = sb;
  }
}

__global__ void sedge_kernel(const float* __restrict__ ea, const float* __restrict__ weh,
                             float* __restrict__ sedge, int E) {
  int e = blockIdx.x * blockDim.x + threadIdx.x;
  if (e >= E) return;
  float a[16];
#pragma unroll
  for (int i = 0; i < 16; ++i) a[i] = ea[(long)e * 16 + i];
#pragma unroll
  for (int h = 0; h < 4; ++h) {
    float s = weh[64 + h];
#pragma unroll
    for (int i = 0; i < 16; ++i) s += a[i] * weh[h * 16 + i];
    sedge[(long)e * 4 + h] = s;
  }
}

__device__ __forceinline__ float leaky02(float l) { return l > 0.f ? l : 0.2f * l; }

__device__ __forceinline__ void atomicMaxF(float* a, float v) {
  unsigned int* p = (unsigned int*)a;
  unsigned int old = *p;
  while (__uint_as_float(old) < v) {
    unsigned int assumed = old;
    old = atomicCAS(p, assumed, __float_as_uint(v));
    if (old == assumed) break;
  }
}

__global__ void att_max_kernel(const float* __restrict__ ssrc, const float* __restrict__ sdst,
    const float* __restrict__ sedge, const int* __restrict__ src, const int* __restrict__ dst,
    const int* __restrict__ edelta, const int* __restrict__ eself, int mode,
    float* __restrict__ m, int E) {
  int e = blockIdx.x * blockDim.x + threadIdx.x;
  if (e >= E) return;
  int s = src[e], d = dst[e];
  bool msk = edge_mask(mode, edelta[e], eself[e]);
#pragma unroll
  for (int h = 0; h < 4; ++h) {
    float l = msk ? leaky02(ssrc[s * 4 + h] + sdst[d * 4 + h] + sedge[(long)e * 4 + h]) : -1e9f;
    atomicMaxF(&m[d * 4 + h], l);
  }
}

__global__ void fix_m_kernel(float* __restrict__ m, int n) {
  int i = blockIdx.x * blockDim.x + threadIdx.x;
  if (i < n && m[i] < -9e29f) m[i] = 0.f;
}

__global__ void att_denom_kernel(const float* __restrict__ ssrc, const float* __restrict__ sdst,
    const float* __restrict__ sedge, const int* __restrict__ src, const int* __restrict__ dst,
    const int* __restrict__ edelta, const int* __restrict__ eself, int mode,
    const float* __restrict__ m, float* __restrict__ denom, int E) {
  int e = blockIdx.x * blockDim.x + threadIdx.x;
  if (e >= E) return;
  int s = src[e], d = dst[e];
  if (!edge_mask(mode, edelta[e], eself[e])) return;
#pragma unroll
  for (int h = 0; h < 4; ++h) {
    float l = leaky02(ssrc[s * 4 + h] + sdst[d * 4 + h] + sedge[(long)e * 4 + h]);
    atomicAdd(&denom[d * 4 + h], __expf(l - m[d * 4 + h]));
  }
}

__global__ void att_alpha_kernel(const float* __restrict__ ssrc, const float* __restrict__ sdst,
    const float* __restrict__ sedge, const int* __restrict__ src, const int* __restrict__ dst,
    const int* __restrict__ edelta, const int* __restrict__ eself, int mode,
    const float* __restrict__ m, const float* __restrict__ denom,
    float* __restrict__ alpha, int E) {
  int e = blockIdx.x * blockDim.x + threadIdx.x;
  if (e >= E) return;
  int s = src[e], d = dst[e];
  bool msk = edge_mask(mode, edelta[e], eself[e]);
#pragma unroll
  for (int h = 0; h < 4; ++h) {
    float a = 0.f;
    if (msk) {
      float l = leaky02(ssrc[s * 4 + h] + sdst[d * 4 + h] + sedge[(long)e * 4 + h]);
      a = __expf(l - m[d * 4 + h]) / fmaxf(denom[d * 4 + h], 1e-16f);
    }
    alpha[(long)e * 4 + h] = a;
  }
}

__global__ void att_aggr_kernel(const float* __restrict__ alpha, const float* __restrict__ xh,
    const int* __restrict__ src, const int* __restrict__ dst,
    float* __restrict__ aggr, int E) {
  long i = (long)blockIdx.x * blockDim.x + threadIdx.x;
  if (i >= (long)E * 64) return;
  int c = (int)(i & 63);
  int e = (int)(i >> 6);
  int s = src[e], d = dst[e];
#pragma unroll
  for (int h = 0; h < 4; ++h) {
    float al = alpha[(long)e * 4 + h];
    if (al != 0.f)
      atomicAdd(&aggr[(long)d * 256 + h * 64 + c], al * xh[(long)s * 256 + h * 64 + c]);
  }
}

__global__ void aggr_mean_kernel(const float* __restrict__ aggr, float* __restrict__ am, int N) {
  long i = (long)blockIdx.x * blockDim.x + threadIdx.x;
  if (i >= (long)N * 64) return;
  long n = i >> 6; int c = (int)(i & 63);
  const float* p = aggr + n * 256 + c;
  am[i] = 0.25f * (p[0] + p[64] + p[128] + p[192]);
}

// ---------------------------------------------------------------------------
// Elementwise / misc
// ---------------------------------------------------------------------------
__global__ void fill_kernel(float* __restrict__ p, float v, long n) {
  long i = (long)blockIdx.x * blockDim.x + threadIdx.x;
  if (i < n) p[i] = v;
}

__global__ void combine_bn_relu_kernel(const float* __restrict__ A, const float* __restrict__ B,
    const float* __restrict__ Cc,
    const float* g, const float* bb, const float* mu, const float* var,
    float* __restrict__ out, int N) {
  long i = (long)blockIdx.x * blockDim.x + threadIdx.x;
  if (i >= (long)N * 64) return;
  int col = (int)(i & 63);
  float v = A[i] + B[i] + (Cc ? Cc[i] : 0.f);
  v = (v - mu[col]) * (g[col] / sqrtf(var[col] + BN_EPS)) + bb[col];
  out[i] = fmaxf(v, 0.f);
}

__global__ void add_kernel(const float* __restrict__ A, const float* __restrict__ B,
                           float* __restrict__ out, long n) {
  long i = (long)blockIdx.x * blockDim.x + threadIdx.x;
  if (i < n) out[i] = A[i] + B[i];
}

__global__ void cnt_kernel(const int* __restrict__ dst, const int* __restrict__ edelta,
                           const int* __restrict__ eself, int mode,
                           float* __restrict__ cnt, int E) {
  int e = blockIdx.x * blockDim.x + threadIdx.x;
  if (e >= E) return;
  if (edge_mask(mode, edelta[e], eself[e])) atomicAdd(&cnt[dst[e]], 1.f);
}

__global__ void ec_finalize_kernel(const float* __restrict__ S, const float* __restrict__ cnt,
                                   float* __restrict__ out, int N) {
  long i = (long)blockIdx.x * blockDim.x + threadIdx.x;
  if (i >= (long)N * 64) return;
  long n = i >> 6;
  out[i] = S[i] / fmaxf(cnt[n], 1.f);
}

__global__ void fc_head_kernel(const float* __restrict__ X, const float* __restrict__ W,
                               const float* __restrict__ b, float* __restrict__ out, int N) {
  int i = blockIdx.x * blockDim.x + threadIdx.x;
  if (i >= N * 2) return;
  int n = i >> 1, j = i & 1;
  float s = b[j];
  for (int c = 0; c < 64; ++c) s += X[(long)n * 64 + c] * W[c * 2 + j];
  out[i] = s;
}

__global__ void fc_av_kernel(const float* __restrict__ gf,
                             const float* __restrict__ wa, const float* __restrict__ ba,
                             const float* __restrict__ wv, const float* __restrict__ bv,
                             float* __restrict__ out, const int* __restrict__ na_ptr, int N) {
  int i = blockIdx.x * blockDim.x + threadIdx.x;
  if (i >= N * 2) return;
  int n = i >> 1, j = i & 1;
  int na = *na_ptr;
  const float* W; const float* B; long off;
  if (n < na) { W = wa; B = ba; off = (long)n * 2 + j; }
  else        { W = wv; B = bv; off = (long)na * 2 + (long)(n - na) * 2 + j; }
  float s = B[j];
  for (int c = 0; c < 64; ++c) s += gf[(long)n * 64 + c] * W[c * 2 + j];
  out[off] = s;
}

// ---------------------------------------------------------------------------
// Host orchestration
// ---------------------------------------------------------------------------
extern "C" void kernel_launch(void* const* d_in, const int* in_sizes, int n_in,
                              void* d_out, int out_size, void* d_ws, size_t ws_size,
                              hipStream_t stream) {
  (void)n_in; (void)out_size; (void)ws_size;
  int idx = 0;
  auto F = [&]() { return (const float*)d_in[idx++]; };
  struct BN { const float *g, *b, *m, *v; };
  struct LPP { BN bn1; const float *f1w, *f1b; BN bn2; const float *f2w, *f2b; };
  struct KM { const float *lin_w, *lin_b, *lin_e_w, *lin_e_b, *att_src, *att_dst, *att_edge, *bias; LPP nn; };
  auto rdBN = [&]() { BN r; r.g = F(); r.b = F(); r.m = F(); r.v = F(); return r; };
  auto rdLPP = [&]() { LPP r; r.bn1 = rdBN(); r.f1w = F(); r.f1b = F(); r.bn2 = rdBN(); r.f2w = F(); r.f2b = F(); return r; };
  auto rdKM = [&]() { KM k; k.lin_w = F(); k.lin_b = F(); k.lin_e_w = F(); k.lin_e_b = F();
                      k.att_src = F(); k.att_dst = F(); k.att_edge = F(); k.bias = F(); k.nn = rdLPP(); return k; };

  const float* x = F();
  const float* edge_attr = F();
  const float *l0a_w = F(), *l0a_b = F(), *l0v_w = F(), *l0v_b = F();
  BN b0a = rdBN(), b0v = rdBN();
  KM km1 = rdKM(), km2 = rdKM();
  BN bn1 = rdBN();
  LPP l2 = rdLPP();
  BN bn2 = rdBN();
  LPP l3 = rdLPP();
  const float *fc_w = F(), *fc_b = F(), *fca_w = F(), *fca_b = F(), *fcv_w = F(), *fcv_b = F();
  const int* edge_index = (const int*)d_in[idx++];
  const int* edelta = (const int*)d_in[idx++];
  const int* eself = (const int*)d_in[idx++];
  const int* na_ptr = (const int*)d_in[idx++];

  const int N = in_sizes[0] / 1024;  // x: [N,2,512]
  const int E = in_sizes[1] / 16;    // edge_attr: [E,16]
  const int* esrc = edge_index;
  const int* edst = edge_index + E;

  // ---- workspace bump allocator ----
  char* wsp = (char*)d_ws;
  size_t off = 0;
  auto alloc = [&](size_t bytes) { size_t a = (off + 255) & ~((size_t)255); void* p = wsp + a; off = a + bytes; return p; };
  auto allocF = [&](long n) { return (float*)alloc((size_t)n * 4); };
  auto allocH = [&](long n) { return (_Float16*)alloc((size_t)n * 2); };

  float* gf = allocF((long)N * 64);
  float* g1b = allocF((long)N * 64);
  float* g2b = allocF((long)N * 64);
  float* pA = allocF((long)N * 64);
  float* pB = allocF((long)N * 64);
  float* pC = allocF((long)N * 64);
  float* xh = allocF((long)N * 256);
  float* aggr = allocF((long)N * 256);
  float* am = allocF((long)N * 64);
  float* ssrc = allocF((long)N * 4);
  float* sdst = allocF((long)N * 4);
  float* mbuf = allocF((long)N * 4);
  float* dbuf = allocF((long)N * 4);
  float* alpha = allocF((long)E * 4);
  float* se1 = allocF((long)E * 4);
  float* se2 = allocF((long)E * 4);
  float* Sbuf = allocF((long)N * 64);
  float* cntb = allocF(N);
  float* weh1 = allocF(68);
  float* weh2 = allocF(68);

  auto packB = [&](const float* W, int K, int Nc) {
    int numKt = (K + 31) / 32, numNt = (Nc + 15) / 16;
    long total = (long)numKt * numNt * 512;
    _Float16* p = allocH(total);
    pack_b_kernel<<<(int)((total + 255) / 256), 256, 0, stream>>>(W, p, K, Nc);
    return p;
  };
  _Float16* Wa_p = packB(l0a_w, 512, 64);
  _Float16* Wv_p = packB(l0v_w, 512, 64);
  _Float16* km1lin_p = packB(km1.lin_w, 64, 256);
  _Float16* km2lin_p = packB(km2.lin_w, 64, 256);
  _Float16* km1f1_p = packB(km1.nn.f1w, 128, 64);
  _Float16* km1f2_p = packB(km1.nn.f2w, 64, 64);
  _Float16* km2f1_p = packB(km2.nn.f1w, 128, 64);
  _Float16* km2f2_p = packB(km2.nn.f2w, 64, 64);
  _Float16* l2f1_p = packB(l2.f1w, 128, 64);
  _Float16* l2f2_p = packB(l2.f2w, 64, 64);
  _Float16* l3f1_p = packB(l3.f1w, 128, 64);
  _Float16* l3f2_p = packB(l3.f2w, 64, 64);

  const int mtN = (N + 15) / 16;
  const int mtE = (E + 15) / 16;
  auto gridE = [&](long n) { return (int)((n + 255) / 256); };

  // ---- layer 0 ----
  gemm_layer0_kernel<<<dim3(mtN, 4), 32, 0, stream>>>(x, 1024, 0, Wa_p, 512, l0a_b,
      b0a.g, b0a.b, b0a.m, b0a.v, gf, na_ptr, 0, N);
  gemm_layer0_kernel<<<dim3(mtN, 4), 32, 0, stream>>>(x, 1024, 512, Wv_p, 512, l0v_b,
      b0v.g, b0v.b, b0v.m, b0v.v, gf, na_ptr, 1, N);

  // ---- s_edge precompute (per KmConv param set) ----
  prep_sedge_kernel<<<1, 64, 0, stream>>>(km1.lin_e_w, km1.lin_e_b, km1.att_edge, weh1);
  sedge_kernel<<<gridE(E), 256, 0, stream>>>(edge_attr, weh1, se1, E);
  prep_sedge_kernel<<<1, 64, 0, stream>>>(km2.lin_e_w, km2.lin_e_b, km2.att_edge, weh2);
  sedge_kernel<<<gridE(E), 256, 0, stream>>>(edge_attr, weh2, se2, E);

  // ---- KmConv stream ----
  auto run_km = [&](const KM& P, const _Float16* linp, const _Float16* f1p, const _Float16* f2p,
                    const float* se, const int* modes, int nm, const float* xin,
                    float* ping, float* pong) -> const float* {
    const float* cur = xin;
    float* outs[2] = {ping, pong};
    for (int mi = 0; mi < nm; ++mi) {
      int mode = modes[mi];
      gemm_bias_kernel<<<dim3(mtN, 16), 32, 0, stream>>>(cur, 64, linp, 64, 256, P.lin_b, xh, N);
      s_nodes_kernel<<<gridE((long)N * 4), 256, 0, stream>>>(xh, P.att_src, P.att_dst, ssrc, sdst, N);
      fill_kernel<<<gridE((long)N * 4), 256, 0, stream>>>(mbuf, -1e30f, (long)N * 4);
      fill_kernel<<<gridE((long)N * 4), 256, 0, stream>>>(dbuf, 0.f, (long)N * 4);
      fill_kernel<<<gridE((long)N * 256), 256, 0, stream>>>(aggr, 0.f, (long)N * 256);
      att_max_kernel<<<gridE(E), 256, 0, stream>>>(ssrc, sdst, se, esrc, edst, edelta, eself, mode, mbuf, E);
      fix_m_kernel<<<gridE((long)N * 4), 256, 0, stream>>>(mbuf, N * 4);
      att_denom_kernel<<<gridE(E), 256, 0, stream>>>(ssrc, sdst, se, esrc, edst, edelta, eself, mode, mbuf, dbuf, E);
      att_alpha_kernel<<<gridE(E), 256, 0, stream>>>(ssrc, sdst, se, esrc, edst, edelta, eself, mode, mbuf, dbuf, alpha, E);
      att_aggr_kernel<<<gridE((long)E * 64), 256, 0, stream>>>(alpha, xh, esrc, edst, aggr, E);
      aggr_mean_kernel<<<gridE((long)N * 64), 256, 0, stream>>>(aggr, am, N);
      km_lpp_kernel<<<mtN, 32, 0, stream>>>(cur, am,
          P.nn.bn1.g, P.nn.bn1.b, P.nn.bn1.m, P.nn.bn1.v, f1p, P.nn.f1b,
          P.nn.bn2.g, P.nn.bn2.b, P.nn.bn2.m, P.nn.bn2.v, f2p, P.nn.f2b,
          P.bias, outs[mi & 1], N);
      cur = outs[mi & 1];
    }
    return cur;
  };

  const int m12[2] = {1, 2};
  const int m345[3] = {3, 4, 5};
  const float* g11 = run_km(km1, km1lin_p, km1f1_p, km1f2_p, se1, m12, 2, gf, pA, pB);   // -> pB
  const float* g12 = run_km(km2, km2lin_p, km2f1_p, km2f2_p, se2, m345, 3, gf, pC, pA);  // -> pC
  combine_bn_relu_kernel<<<gridE((long)N * 64), 256, 0, stream>>>(g11, g12, gf,
      bn1.g, bn1.b, bn1.m, bn1.v, g1b, N);

  // ---- EdgeConv streams ----
  auto run_ec = [&](const LPP& P, const _Float16* f1p, const _Float16* f2p,
                    const int* modes, int nm, const float* xin,
                    float* ping, float* pong) -> const float* {
    const float* cur = xin;
    float* outs[2] = {ping, pong};
    for (int mi = 0; mi < nm; ++mi) {
      int mode = modes[mi];
      fill_kernel<<<gridE((long)N * 64), 256, 0, stream>>>(Sbuf, 0.f, (long)N * 64);
      fill_kernel<<<gridE(N), 256, 0, stream>>>(cntb, 0.f, N);
      ec_lpp_kernel<<<mtE, 32, 0, stream>>>(cur, esrc, edst, edelta, eself, mode,
          P.bn1.g, P.bn1.b, P.bn1.m, P.bn1.v, f1p, P.f1b,
          P.bn2.g, P.bn2.b, P.bn2.m, P.bn2.v, f2p, P.f2b, Sbuf, E);
      cnt_kernel<<<gridE(E), 256, 0, stream>>>(edst, edelta, eself, mode, cntb, E);
      ec_finalize_kernel<<<gridE((long)N * 64), 256, 0, stream>>>(Sbuf, cntb, outs[mi & 1], N);
      cur = outs[mi & 1];
    }
    return cur;
  };

  const float* e1 = run_ec(l2, l2f1_p, l2f2_p, m12, 2, g1b, pA, pB);  // -> pB
  combine_bn_relu_kernel<<<gridE((long)N * 64), 256, 0, stream>>>(e1, g1b, nullptr,
      bn2.g, bn2.b, bn2.m, bn2.v, g2b, N);
  const float* e2 = run_ec(l3, l3f1_p, l3f2_p, m12, 2, g2b, pA, pB);  // -> pB
  add_kernel<<<gridE((long)N * 64), 256, 0, stream>>>(e2, g2b, pA, (long)N * 64);  // g3 -> pA

  // ---- heads ----
  fc_head_kernel<<<gridE((long)N * 2), 256, 0, stream>>>(pA, fc_w, fc_b, (float*)d_out, N);
  fc_av_kernel<<<gridE((long)N * 2), 256, 0, stream>>>(gf, fca_w, fca_b, fcv_w, fcv_b,
      (float*)d_out + (long)N * 2, na_ptr, N);
}
